// STFT_67860483277597
// MI455X (gfx1250) — compile-verified, matrix operand order checked by reading, not run on hardware
//
#include <hip/hip_runtime.h>
#include <hip/hip_bf16.h>
#include <math.h>

typedef __attribute__((ext_vector_type(16))) _Float16 v16h;
typedef __attribute__((ext_vector_type(8)))  _Float16 v8h;
typedef __attribute__((ext_vector_type(8)))  float    v8f;

#define FILTER_LEN   1024
#define HOP          512
#define CUTOFF       513           // real rows / imag rows
#define PAD          512
#define BATCH        32
#define NSAMP        524288
#define PADLEN       (NSAMP + 2 * PAD)     // 525312
#define NFRAMES      1025
#define FREQ_PAD     544           // 513 padded up to multiple of 32
#define WAVE_FRAMES  32            // 2 frame tiles of 16 per wave
#define WAVES_PER_BLK 8
#define BLK_FRAMES   (WAVE_FRAMES * WAVES_PER_BLK)   // 256
#define BLK_FREQ     32            // 2 freq tiles of 16 per wave (shared by block)

// ---------------------------------------------------------------------------
// Prep kernel 1: reflect-pad the signal and convert f32 -> f16.
// One element per thread; total BATCH*PADLEN = 16,809,984 (divisible by 256).
// ---------------------------------------------------------------------------
__global__ __launch_bounds__(256) void stft_pad_f16(
    const float* __restrict__ in, _Float16* __restrict__ sig)
{
    const long long idx = (long long)blockIdx.x * 256 + threadIdx.x;
    const int b = (int)(idx / PADLEN);
    const int i = (int)(idx % PADLEN);
    int j = i - PAD;
    if (j < 0)            j = -j;                    // reflect (exclude edge)
    else if (j >= NSAMP)  j = 2 * NSAMP - 2 - j;
    sig[idx] = (_Float16)in[(long long)b * NSAMP + j];
}

// ---------------------------------------------------------------------------
// Prep kernel 2: split basis into Real/Imag f16 matrices, zero-pad rows to 544.
// Total FREQ_PAD*FILTER_LEN = 557,056 (divisible by 256 -> 2176 blocks).
// ---------------------------------------------------------------------------
__global__ __launch_bounds__(256) void stft_basis_f16(
    const float* __restrict__ basis,
    _Float16* __restrict__ Rb, _Float16* __restrict__ Ib)
{
    const int idx = blockIdx.x * 256 + threadIdx.x;
    const int row = idx >> 10;          // / FILTER_LEN
    const int k   = idx & 1023;         // % FILTER_LEN
    float rv = 0.f, iv = 0.f;
    if (row < CUTOFF) {
        rv = basis[(long long)row * FILTER_LEN + k];
        iv = basis[(long long)(row + CUTOFF) * FILTER_LEN + k];
    }
    Rb[idx] = (_Float16)rv;
    Ib[idx] = (_Float16)iv;
}

// ---------------------------------------------------------------------------
// Fragment load matching the CDNA5 16-bit A/B VGPR layout:
//   lanes 0-15  : elements 0..7 = K[0..7],   8..15 = K[16..23]
//   lanes 16-31 : elements 0..7 = K[8..15],  8..15 = K[24..31]
// `p` already includes the per-lane row offset and koff; two b128 loads.
// ---------------------------------------------------------------------------
__device__ __forceinline__ v16h load_frag(const _Float16* __restrict__ p)
{
    v8h lo = *(const v8h*)(p);
    v8h hi = *(const v8h*)(p + 16);
    v16h r;
#pragma unroll
    for (int i = 0; i < 8; ++i) { r[i] = lo[i]; r[i + 8] = hi[i]; }
    return r;
}

#define WMMA_F16(A, B, C) \
    __builtin_amdgcn_wmma_f32_16x16x32_f16(false, (A), false, (B), (short)0, (C), false, false)

// ---------------------------------------------------------------------------
// Main GEMM + magnitude kernel, 2x2 register tiling per wave.
// Grid: x = frame tiles of 256, y = 17 freq tiles of 32, z = batch.
// Each wave: 32(freq) x 32(frame) output = 4 tiles x (real+imag)
//            = 8 f32 accumulators; per K-step 12 b128 loads feed 8 WMMAs.
// ---------------------------------------------------------------------------
__global__ __launch_bounds__(256) void stft_wmma_kernel(
    const _Float16* __restrict__ sig,
    const _Float16* __restrict__ Rb,
    const _Float16* __restrict__ Ib,
    float* __restrict__ out)
{
    const int lane = threadIdx.x & 31;
    const int wave = threadIdx.x >> 5;

    const int n_base = blockIdx.x * BLK_FRAMES + wave * WAVE_FRAMES; // frame base
    const int m_base = blockIdx.y * BLK_FREQ;                        // freq base
    const int b      = blockIdx.z;

    const int r    = lane & 15;          // row-in-tile (freq for A, frame for B)
    const int koff = (lane >> 4) * 8;    // lane-half K offset

    // B columns: frame n = contiguous samples at n*HOP (stride 512); clamp loads
    int f0 = n_base + r;           if (f0 > NFRAMES - 1) f0 = NFRAMES - 1;
    int f1 = n_base + 16 + r;      if (f1 > NFRAMES - 1) f1 = NFRAMES - 1;
    const _Float16* sigb = sig + (long long)b * PADLEN + koff;
    const _Float16* sb0 = sigb + (long long)f0 * HOP;
    const _Float16* sb1 = sigb + (long long)f1 * HOP;

    const _Float16* ra0 = Rb + (long long)(m_base + r)      * FILTER_LEN + koff;
    const _Float16* ra1 = Rb + (long long)(m_base + 16 + r) * FILTER_LEN + koff;
    const _Float16* ia0 = Ib + (long long)(m_base + r)      * FILTER_LEN + koff;
    const _Float16* ia1 = Ib + (long long)(m_base + 16 + r) * FILTER_LEN + koff;

    v8f aR00 = {}, aR01 = {}, aR10 = {}, aR11 = {};
    v8f aI00 = {}, aI01 = {}, aI10 = {}, aI11 = {};

#pragma unroll 2
    for (int k0 = 0; k0 < FILTER_LEN; k0 += 32) {
        v16h B0  = load_frag(sb0 + k0);
        v16h B1  = load_frag(sb1 + k0);
        v16h Ar0 = load_frag(ra0 + k0);
        v16h Ai0 = load_frag(ia0 + k0);
        v16h Ar1 = load_frag(ra1 + k0);
        v16h Ai1 = load_frag(ia1 + k0);

        aR00 = WMMA_F16(Ar0, B0, aR00);
        aR01 = WMMA_F16(Ar0, B1, aR01);
        aI00 = WMMA_F16(Ai0, B0, aI00);
        aI01 = WMMA_F16(Ai0, B1, aI01);
        aR10 = WMMA_F16(Ar1, B0, aR10);
        aR11 = WMMA_F16(Ar1, B1, aR11);
        aI10 = WMMA_F16(Ai1, B0, aI10);
        aI11 = WMMA_F16(Ai1, B1, aI11);
    }

    // Epilogue: magnitude, 16x16 f32 C layout:
    //   lanes 0-15: N=lane,     VGPR v -> M=v
    //   lanes 16-31: N=lane-16, VGPR v -> M=v+8
    const int ncol  = lane & 15;
    const int mhalf = (lane >> 4) * 8;

    const v8f* accR[2][2] = { { &aR00, &aR01 }, { &aR10, &aR11 } };
    const v8f* accI[2][2] = { { &aI00, &aI01 }, { &aI10, &aI11 } };

#pragma unroll
    for (int mt = 0; mt < 2; ++mt) {
#pragma unroll
        for (int nt = 0; nt < 2; ++nt) {
            const int n = n_base + nt * 16 + ncol;
            if (n < NFRAMES) {
                const v8f vr = *accR[mt][nt];
                const v8f vi = *accI[mt][nt];
#pragma unroll
                for (int v = 0; v < 8; ++v) {
                    const int f = m_base + mt * 16 + v + mhalf;
                    if (f < CUTOFF) {
                        const float xr = vr[v];
                        const float xi = vi[v];
                        out[((long long)b * CUTOFF + f) * NFRAMES + n] =
                            sqrtf(xr * xr + xi * xi);
                    }
                }
            }
        }
    }
}

// ---------------------------------------------------------------------------
// Launch
// ---------------------------------------------------------------------------
extern "C" void kernel_launch(void* const* d_in, const int* in_sizes, int n_in,
                              void* d_out, int out_size, void* d_ws, size_t ws_size,
                              hipStream_t stream)
{
    const float* input = (const float*)d_in[0];   // [32, 524288] f32
    const float* basis = (const float*)d_in[1];   // [1026, 1024] f32
    float* out = (float*)d_out;                   // [32, 513, 1025] f32

    // Workspace layout (256B-aligned sizes):
    //   sig_f16 : BATCH*PADLEN halves     = 33,619,968 B
    //   Rb      : FREQ_PAD*FILTER_LEN h   =  1,114,112 B
    //   Ib      : FREQ_PAD*FILTER_LEN h   =  1,114,112 B
    _Float16* sig = (_Float16*)d_ws;
    _Float16* Rb  = (_Float16*)((char*)d_ws + (size_t)BATCH * PADLEN * sizeof(_Float16));
    _Float16* Ib  = Rb + (size_t)FREQ_PAD * FILTER_LEN;

    // 1) reflect-pad + f32->f16   (16,809,984 / 256 = 65,664 blocks)
    stft_pad_f16<<<(BATCH * PADLEN) / 256, 256, 0, stream>>>(input, sig);

    // 2) basis split/convert      (557,056 / 256 = 2,176 blocks)
    stft_basis_f16<<<(FREQ_PAD * FILTER_LEN) / 256, 256, 0, stream>>>(basis, Rb, Ib);

    // 3) WMMA GEMM + magnitude
    dim3 grid((NFRAMES + BLK_FRAMES - 1) / BLK_FRAMES,   // 5
              FREQ_PAD / BLK_FREQ,                        // 17
              BATCH);                                     // 32
    stft_wmma_kernel<<<grid, 256, 0, stream>>>(sig, Rb, Ib, out);
}
